// TransformerClassifier_27642409517146
// MI455X (gfx1250) — matrix operand order repaired and checked by reading, not compiled, not run on hardware
//
#include <hip/hip_runtime.h>

typedef __attribute__((ext_vector_type(16))) __bf16 v16bf;
typedef __attribute__((ext_vector_type(8)))  float  v8f;

union BFragU {
  v16bf bf;
  __bf16 h[16];
  unsigned short us[16];
  uint4 q[2];
};

__device__ __forceinline__ unsigned short f2bf_bits(float f) {
  union { __bf16 b; unsigned short u; } cv;
  cv.b = (__bf16)f;
  return cv.u;
}

// fast sigmoid: v_exp_f32 + v_rcp_f32 (avoids IEEE div expansion)
__device__ __forceinline__ float sigmoidf_(float x) {
  return __builtin_amdgcn_rcpf(1.0f + __expf(-x));
}

// sum across each 16-lane row group using DPP16 ROW_XMASK (pure VALU, no LDS)
template <int MASK>
__device__ __forceinline__ float dpp_xadd(float x) {
  int p = __builtin_amdgcn_update_dpp(0, __float_as_int(x), 0x160 | MASK, 0xF, 0xF, true);
  return x + __int_as_float(p);
}
__device__ __forceinline__ float row16_sum(float x) {
  x = dpp_xadd<1>(x);
  x = dpp_xadd<2>(x);
  x = dpp_xadd<4>(x);
  x = dpp_xadd<8>(x);
  return x;
}

#define WMMA_BF16(A, Bv, C) \
  __builtin_amdgcn_wmma_f32_16x16x32_bf16(false, (A), false, (Bv), (short)0, (C), false, false)

// A fragment (16x32 bf16) from an (m,k)->float accessor, wave32 layout:
// lane<16 : K = 0..15; lane>=16 : K = 16..31 (contiguous); M = lane&15.
template <typename F>
__device__ __forceinline__ v16bf make_afrag(int lane, F getx) {
  const int m  = lane & 15;
  const int kb = (lane >= 16) ? 16 : 0;
  BFragU u;
#pragma unroll
  for (int e = 0; e < 8; ++e) {
    u.h[e]     = (__bf16)getx(m, kb + e);
    u.h[8 + e] = (__bf16)getx(m, kb + 8 + e);
  }
  return u.bf;
}

// B fragment (32x16 bf16) from pre-transposed bf16 weights WT[N][K]:
// lane holds column N = N0+(lane&15), rows K = K0 + (lane>=16?16:0) + 0..15 (contiguous).
__device__ __forceinline__ v16bf load_bfrag(const unsigned short* __restrict__ WT,
                                            int ldK, int N0, int K0, int lane) {
  const int n = N0 + (lane & 15);
  const int k = K0 + ((lane >= 16) ? 16 : 0);
  const uint4* p = (const uint4*)(WT + (size_t)n * ldK + k);
  BFragU u;
  u.q[0] = p[0];
  u.q[1] = p[1];
  return u.bf;
}

// One 16-item tile of the score MLP:  s(x) = sigmoid(x@w1+b1)@w2 + b2,  x in R^32.
template <typename F>
__device__ __forceinline__ void score_tile(int lane, F getx, v16bf b0, v16bf b1v,
                                           const float* __restrict__ sb1,
                                           const float* __restrict__ sw2, float sb2,
                                           float* __restrict__ dst, int i0) {
  v16bf a = make_afrag(lane, getx);
  v8f c0, c1;
  const float bias0 = sb1[lane & 15];
  const float bias1 = sb1[16 + (lane & 15)];
#pragma unroll
  for (int i = 0; i < 8; ++i) { c0[i] = bias0; c1[i] = bias1; }
  c0 = WMMA_BF16(a, b0, c0);
  c1 = WMMA_BF16(a, b1v, c1);
  const float w2lo = sw2[lane & 15];
  const float w2hi = sw2[16 + (lane & 15)];
#pragma unroll
  for (int rv = 0; rv < 8; ++rv) {
    // D(M,N): M = rv + 8*(lane>=16) constant per half-wave, N = lane&15
    float t = sigmoidf_(c0[rv]) * w2lo + sigmoidf_(c1[rv]) * w2hi;
    t = row16_sum(t);                 // DPP16 ROW_XMASK reduction, no LDS traffic
    if ((lane & 15) == 0)
      dst[i0 + rv + ((lane >= 16) ? 8 : 0)] = t + sb2;
  }
}

__global__ void transpose_bf16_k(const float* __restrict__ src,
                                 unsigned short* __restrict__ dst, int R, int C) {
  int i = blockIdx.x * blockDim.x + threadIdx.x;
  if (i < R * C) {
    int r = i / C, c = i - r * C;
    dst[c * R + r] = f2bf_bits(src[i]);   // dst is [C][R] = W^T, bf16
  }
}

#define SCALE_F 0.17677669529663687f   // 1/sqrt(32)

// LDS layout (floats); regions aliased across phases:
//  OFF_EMB : emb[key][row][256]              (phases 1..7a)
//  OFF_B   : Latt[16][256] | Satt[16][256]   (3..5)  -> zatt[16][256] | zd[16][256] (6..8)
//  OFF_ATT : att[16][256]                    (5..6)
//  OFF_S   : scoreL[768]+scoreS[256] (2..3)  -> hA[1024]+hB[1024] (7)
//  OFF_RL/RS : rl/rs[16][8]                  (4..5)
//  OFF_Z0  : z0[16][6]
#define OFF_EMB 0
#define OFF_B   24576
#define OFF_ATT 32768
#define OFF_S   36864
#define OFF_RL  38912
#define OFF_RS  39040
#define OFF_Z0  39168
#define SMEM_FLOATS 39264

__launch_bounds__(256)
__global__ void fused_txc(
    const float* __restrict__ age, const float* __restrict__ gender,
    const int* __restrict__ mtong, const float* __restrict__ mood,
    const float* __restrict__ gen1, const float* __restrict__ gen2,
    const float* __restrict__ mt_table,
    const float* __restrict__ emb_w, const float* __restrict__ emb_b,
    const float* __restrict__ ln_g, const float* __restrict__ ln_b,
    const float* __restrict__ score_b1, const float* __restrict__ score_w2,
    const float* __restrict__ score_b2,
    const float* __restrict__ mha_b,
    const float* __restrict__ dnn_b_in, const float* __restrict__ dnn_bh,
    const float* __restrict__ dnn_b_out,
    const float* __restrict__ W_zd, const float* __restrict__ W_z0,
    const float* __restrict__ W_zatt, const float* __restrict__ out_bias,
    const unsigned short* __restrict__ w1T,
    const unsigned short* __restrict__ mha_wT,
    const unsigned short* __restrict__ w_inT,
    const unsigned short* __restrict__ whT,
    const unsigned short* __restrict__ w_outT,
    float* __restrict__ out, int B) {
  extern __shared__ float smem[];
  const int tid = threadIdx.x;
  const int wave = tid >> 5;
  const int lane = tid & 31;
  const int rowBase = blockIdx.x * 16;

  if (tid == 0) {
    __builtin_prefetch(mha_wT, 0, 0);
    __builtin_prefetch(w_inT, 0, 0);
  }

  // ---- Phase 0: gather per-row scalars -> z0 ----
  if (tid < 16) {
    int row = rowBase + tid;
    float v0 = 0.f, v1 = 0.f, v2 = 0.f, v3 = 0.f, v4 = 0.f, v5 = 0.f;
    if (row < B) {
      v0 = age[row]; v1 = gender[row];
      int mi = mtong[row];
      mi = (mi < 0) ? 0 : ((mi > 39) ? 39 : mi);
      v2 = mt_table[mi];
      v3 = mood[row]; v4 = gen1[row]; v5 = gen2[row];
    }
    float* z = &smem[OFF_Z0 + tid * 6];
    z[0] = v0; z[1] = v1; z[2] = v2; z[3] = v3; z[4] = v4; z[5] = v5;
  }
  __syncthreads();

  // ---- Phase 1: lin = z*w+b, LayerNorm over 256 -> emb (fp32 in LDS) ----
  if (tid < 96) {
    const int k = tid / 16, r = tid % 16;
    const float z = smem[OFF_Z0 + r * 6 + k];
    const float* ew = emb_w + k * 256;
    const float* eb = emb_b + k * 256;
    float s = 0.f, s2 = 0.f;
    for (int c = 0; c < 256; ++c) {
      float v = fmaf(z, ew[c], eb[c]);
      s += v; s2 += v * v;
    }
    const float mu = s * (1.f / 256.f);
    const float var = s2 * (1.f / 256.f) - mu * mu;
    const float inv = rsqrtf(var + 1e-12f);
    const float* lg = ln_g + k * 256;
    const float* lb = ln_b + k * 256;
    float* dst = &smem[OFF_EMB + (k * 16 + r) * 256];
    for (int c = 0; c < 256; ++c) {
      float v = fmaf(z, ew[c], eb[c]);
      dst[c] = fmaf((v - mu) * inv, lg[c], lb[c]);
    }
  }
  __syncthreads();

  // ---- Phase 2: score MLPs for L (48 tiles) and S (16 tiles), WMMA ----
  {
    v16bf b0 = load_bfrag(w1T, 32, 0, 0, lane);
    v16bf b1v = load_bfrag(w1T, 32, 16, 0, lane);
    for (int T = wave; T < 48; T += 8) {   // 6 tiles per wave, uniform
      const int i0 = T * 16;
      score_tile(lane, [&](int m, int k) -> float {
          int i = i0 + m;
          int r = i / 48; int rem = i - r * 48;
          int cc = rem >> 3, head = rem & 7;
          int u = cc >> 1, g = cc & 1;
          int c = (head << 5) + k;
          return smem[OFF_EMB + (u * 16 + r) * 256 + c] *
                 smem[OFF_EMB + ((4 + g) * 16 + r) * 256 + c] * SCALE_F;
        }, b0, b1v, score_b1, score_w2, score_b2[0], &smem[OFF_S], i0);
    }
    v16bf s0 = load_bfrag(w1T + 1024, 32, 0, 0, lane);
    v16bf s1 = load_bfrag(w1T + 1024, 32, 16, 0, lane);
    for (int T = wave; T < 16; T += 8) {   // 2 tiles per wave
      const int i0 = T * 16;
      score_tile(lane, [&](int m, int k) -> float {
          int i = i0 + m;
          int r = i >> 4; int rem = i & 15;
          int g = rem >> 3, head = rem & 7;
          int c = (head << 5) + k;
          return smem[OFF_EMB + (3 * 16 + r) * 256 + c] *
                 smem[OFF_EMB + ((4 + g) * 16 + r) * 256 + c] * SCALE_F;
        }, s0, s1, score_b1 + 32, score_w2 + 32, score_b2[1],
        &smem[OFF_S + 768], i0);
    }
  }
  __syncthreads();

  // ---- Phase 3: L_att / S_att (weighted sums over combos) ----
  for (int idx = tid; idx < 4096; idx += 256) {
    const int r = idx >> 8, c = idx & 255, head = c >> 5;
    const float geA = smem[OFF_EMB + (4 * 16 + r) * 256 + c];
    const float geB = smem[OFF_EMB + (5 * 16 + r) * 256 + c];
    float la = 0.f;
#pragma unroll
    for (int u = 0; u < 3; ++u) {
      const float ue = smem[OFF_EMB + (u * 16 + r) * 256 + c];
      la = fmaf(smem[OFF_S + r * 48 + (u * 2 + 0) * 8 + head], ue * geA * SCALE_F, la);
      la = fmaf(smem[OFF_S + r * 48 + (u * 2 + 1) * 8 + head], ue * geB * SCALE_F, la);
    }
    const float me = smem[OFF_EMB + (3 * 16 + r) * 256 + c];
    float sa = smem[OFF_S + 768 + r * 16 + head] * (me * geA * SCALE_F) +
               smem[OFF_S + 768 + r * 16 + 8 + head] * (me * geB * SCALE_F);
    smem[OFF_B + r * 256 + c] = la;          // L_att
    smem[OFF_B + 4096 + r * 256 + c] = sa;   // S_att
  }
  __syncthreads();

  // ---- Phase 4: rl / rs scores (one 16-item tile per wave each) ----
  {
    const int i0 = wave * 16;
    {
      v16bf b0 = load_bfrag(w1T + 2048, 32, 0, 0, lane);
      v16bf b1v = load_bfrag(w1T + 2048, 32, 16, 0, lane);
      score_tile(lane, [&](int m, int k) -> float {
          int i = i0 + m; int r = i >> 3, head = i & 7;
          return smem[OFF_B + r * 256 + (head << 5) + k];
        }, b0, b1v, score_b1 + 64, score_w2 + 64, score_b2[2], &smem[OFF_RL], i0);
    }
    {
      v16bf b0 = load_bfrag(w1T + 3072, 32, 0, 0, lane);
      v16bf b1v = load_bfrag(w1T + 3072, 32, 16, 0, lane);
      score_tile(lane, [&](int m, int k) -> float {
          int i = i0 + m; int r = i >> 3, head = i & 7;
          return smem[OFF_B + 4096 + r * 256 + (head << 5) + k];
        }, b0, b1v, score_b1 + 96, score_w2 + 96, score_b2[3], &smem[OFF_RS], i0);
    }
  }
  __syncthreads();

  // ---- Phase 5: softmax over {rl,rs}, att = r0*L_att + r1*S_att ----
  for (int idx = tid; idx < 4096; idx += 256) {
    const int r = idx >> 8, c = idx & 255, head = c >> 5;
    const float a = smem[OFF_RL + r * 8 + head];
    const float b = smem[OFF_RS + r * 8 + head];
    const float mx = fmaxf(a, b);
    const float e0 = __expf(a - mx), e1 = __expf(b - mx);
    const float r0 = e0 * __builtin_amdgcn_rcpf(e0 + e1);
    smem[OFF_ATT + r * 256 + c] =
        r0 * smem[OFF_B + r * 256 + c] + (1.f - r0) * smem[OFF_B + 4096 + r * 256 + c];
  }
  __syncthreads();

  // ---- Phase 6: zatt = att @ mha_w + mha_b  (16x256x256 WMMA) ----
#pragma unroll
  for (int tsel = 0; tsel < 2; ++tsel) {
    const int nt = wave * 2 + tsel;
    v8f acc;
    const float bias = mha_b[nt * 16 + (lane & 15)];
#pragma unroll
    for (int i = 0; i < 8; ++i) acc[i] = bias;
    for (int kk = 0; kk < 8; ++kk) {
      v16bf a = make_afrag(lane, [&](int m, int k) -> float {
          return smem[OFF_ATT + m * 256 + (kk << 5) + k]; });
      v16bf b = load_bfrag(mha_wT, 256, nt * 16, kk * 32, lane);
      acc = WMMA_BF16(a, b, acc);
    }
#pragma unroll
    for (int rv = 0; rv < 8; ++rv) {
      const int row = rv + ((lane >= 16) ? 8 : 0);
      smem[OFF_B + row * 256 + nt * 16 + (lane & 15)] = acc[rv];   // zatt
    }
  }
  __syncthreads();

  // ---- Phase 7a: h = relu(emb_flat(1536) @ dnn_w_in + b)  (waves 0-3) ----
  if (wave < 4) {
    const int nt = wave;
    v8f acc;
    const float bias = dnn_b_in[nt * 16 + (lane & 15)];
#pragma unroll
    for (int i = 0; i < 8; ++i) acc[i] = bias;
    for (int kk = 0; kk < 48; ++kk) {
      v16bf a = make_afrag(lane, [&](int m, int k) -> float {
          int kg = (kk << 5) + k;
          return smem[OFF_EMB + ((kg >> 8) * 16 + m) * 256 + (kg & 255)]; });
      v16bf b = load_bfrag(w_inT, 1536, nt * 16, kk * 32, lane);
      acc = WMMA_BF16(a, b, acc);
    }
#pragma unroll
    for (int rv = 0; rv < 8; ++rv) {
      const int row = rv + ((lane >= 16) ? 8 : 0);
      smem[OFF_S + row * 64 + nt * 16 + (lane & 15)] = fmaxf(acc[rv], 0.f);
    }
  }
  __syncthreads();

  // ---- Phase 7b: two hidden 64x64 relu layers (ping-pong in LDS) ----
  int srcOff = OFF_S, dstOff = OFF_S + 1024;
  for (int layer = 0; layer < 2; ++layer) {
    if (wave < 4) {
      const int nt = wave;
      v8f acc;
      const float bias = dnn_bh[layer * 64 + nt * 16 + (lane & 15)];
#pragma unroll
      for (int i = 0; i < 8; ++i) acc[i] = bias;
#pragma unroll
      for (int kk = 0; kk < 2; ++kk) {
        v16bf a = make_afrag(lane, [&](int m, int k) -> float {
            return smem[srcOff + m * 64 + (kk << 5) + k]; });
        v16bf b = load_bfrag(whT + layer * 4096, 64, nt * 16, kk * 32, lane);
        acc = WMMA_BF16(a, b, acc);
      }
#pragma unroll
      for (int rv = 0; rv < 8; ++rv) {
        const int row = rv + ((lane >= 16) ? 8 : 0);
        smem[dstOff + row * 64 + nt * 16 + (lane & 15)] = fmaxf(acc[rv], 0.f);
      }
    }
    __syncthreads();
    int t = srcOff; srcOff = dstOff; dstOff = t;
  }
  // final h now at srcOff == OFF_S

  // ---- Phase 7c: zd = h @ dnn_w_out + b  (all 8 waves) ----
#pragma unroll
  for (int tsel = 0; tsel < 2; ++tsel) {
    const int nt = wave * 2 + tsel;
    v8f acc;
    const float bias = dnn_b_out[nt * 16 + (lane & 15)];
#pragma unroll
    for (int i = 0; i < 8; ++i) acc[i] = bias;
#pragma unroll
    for (int kk = 0; kk < 2; ++kk) {
      v16bf a = make_afrag(lane, [&](int m, int k) -> float {
          return smem[OFF_S + m * 64 + (kk << 5) + k]; });
      v16bf b = load_bfrag(w_outT, 64, nt * 16, kk * 32, lane);
      acc = WMMA_BF16(a, b, acc);
    }
#pragma unroll
    for (int rv = 0; rv < 8; ++rv) {
      const int row = rv + ((lane >= 16) ? 8 : 0);
      smem[OFF_B + 4096 + row * 256 + nt * 16 + (lane & 15)] = acc[rv];  // zd
    }
  }
  __syncthreads();

  // ---- Phase 8: logits = sigmoid(zd@W_zd + z0@W_z0 + zatt@W_zatt + bias) ----
  if (tid < 160) {
    const int r = tid / 10, o = tid - r * 10;
    const int row = rowBase + r;
    if (row < B) {
      float acc = out_bias[o];
      const float* zd = &smem[OFF_B + 4096 + r * 256];
      const float* za = &smem[OFF_B + r * 256];
      for (int c = 0; c < 256; ++c) acc = fmaf(zd[c], W_zd[c * 10 + o], acc);
#pragma unroll
      for (int k = 0; k < 6; ++k)
        acc = fmaf(smem[OFF_Z0 + r * 6 + k], W_z0[k * 10 + o], acc);
      for (int c = 0; c < 256; ++c) acc = fmaf(za[c], W_zatt[c * 10 + o], acc);
      out[row * 10 + o] = sigmoidf_(acc);
    }
  }
}

extern "C" void kernel_launch(void* const* d_in, const int* in_sizes, int n_in,
                              void* d_out, int out_size, void* d_ws, size_t ws_size,
                              hipStream_t stream) {
  (void)n_in; (void)out_size; (void)ws_size;
  const float* age      = (const float*)d_in[0];
  const float* gender   = (const float*)d_in[1];
  const int*   mtong    = (const int*)  d_in[2];
  const float* mood     = (const float*)d_in[3];
  const float* gen1     = (const float*)d_in[4];
  const float* gen2     = (const float*)d_in[5];
  const float* mt_tab   = (const float*)d_in[6];
  const float* emb_w    = (const float*)d_in[7];
  const float* emb_b    = (const float*)d_in[8];
  const float* ln_g     = (const float*)d_in[9];
  const float* ln_b     = (const float*)d_in[10];
  const float* score_w1 = (const float*)d_in[11];
  const float* score_b1 = (const float*)d_in[12];
  const float* score_w2 = (const float*)d_in[13];
  const float* score_b2 = (const float*)d_in[14];
  const float* mha_w    = (const float*)d_in[15];
  const float* mha_b    = (const float*)d_in[16];
  const float* dnn_w_in = (const float*)d_in[17];
  const float* dnn_b_in = (const float*)d_in[18];
  const float* dnn_wh   = (const float*)d_in[19];
  const float* dnn_bh   = (const float*)d_in[20];
  const float* dnn_w_out= (const float*)d_in[21];
  const float* dnn_b_out= (const float*)d_in[22];
  const float* W_zd     = (const float*)d_in[23];
  const float* W_z0     = (const float*)d_in[24];
  const float* W_zatt   = (const float*)d_in[25];
  const float* out_bias = (const float*)d_in[26];

  const int B = in_sizes[0];

  // Workspace: pre-transposed bf16 weights [N][K] for WMMA B fragments.
  unsigned short* ws    = (unsigned short*)d_ws;
  unsigned short* w1T   = ws;            //  4 * 32*32          = 4096
  unsigned short* mhaT  = ws + 4096;     //  256*256            = 65536
  unsigned short* winT  = ws + 69632;    //  64*1536            = 98304
  unsigned short* whT   = ws + 167936;   //  2 * 64*64          = 8192
  unsigned short* woutT = ws + 176128;   //  256*64             = 16384  (total 385 KB)

  for (int s = 0; s < 4; ++s)
    transpose_bf16_k<<<4, 256, 0, stream>>>(score_w1 + s * 1024, w1T + s * 1024, 32, 32);
  transpose_bf16_k<<<256, 256, 0, stream>>>(mha_w, mhaT, 256, 256);
  transpose_bf16_k<<<384, 256, 0, stream>>>(dnn_w_in, winT, 1536, 64);
  for (int i = 0; i < 2; ++i)
    transpose_bf16_k<<<16, 256, 0, stream>>>(dnn_wh + i * 4096, whT + i * 4096, 64, 64);
  transpose_bf16_k<<<64, 256, 0, stream>>>(dnn_w_out, woutT, 64, 256);

  const int grid = (B + 15) / 16;
  const size_t smemBytes = SMEM_FLOATS * sizeof(float);   // ~153 KB of the 320 KB/WGP
  fused_txc<<<grid, 256, smemBytes, stream>>>(
      age, gender, mtong, mood, gen1, gen2, mt_tab,
      emb_w, emb_b, ln_g, ln_b,
      score_b1, score_w2, score_b2,
      mha_b, dnn_b_in, dnn_bh, dnn_b_out,
      W_zd, W_z0, W_zatt, out_bias,
      w1T, mhaT, winT, whT, woutT,
      (float*)d_out, B);
}
